// GAT_52699248722375
// MI455X (gfx1250) — compile-verified, matrix operand order checked by reading, not run on hardware
//
#include <hip/hip_runtime.h>
#include <hip/hip_bf16.h>
#include <stdint.h>

typedef __attribute__((ext_vector_type(16))) _Float16 v16h;
typedef __attribute__((ext_vector_type(8)))  float    v8f;

#define NEG_SLOPE 0.2f

// ---------- float <-> order-preserving uint (for float atomicMax) ----------
__device__ __forceinline__ unsigned mapOrdered(float f) {
  unsigned u = __float_as_uint(f);
  return (u & 0x80000000u) ? ~u : (u | 0x80000000u);
}
__device__ __forceinline__ float unmapOrdered(unsigned v) {
  return __uint_as_float((v & 0x80000000u) ? (v ^ 0x80000000u) : ~v);
}
// mapOrdered(-inf) == ~0xFF800000u == 0x007FFFFFu
#define MAPPED_NEG_INF 0x007FFFFFu

// ---------------------------- fill scratch -------------------------------
__global__ void fill_u32_kernel(unsigned* __restrict__ p, unsigned v, long long n) {
  long long i = (long long)blockIdx.x * blockDim.x + threadIdx.x;
  if (i < n) p[i] = v;
}

// ---- pack B[K,Nc] (f32 row-major) into f16 WMMA fragment layout ----------
// Bp[((tn*nSlab + ks)*32 + lane)*16 + i] = f16(B[k, col]) with
//   col = tn*16 + lane%16 (clamped), k = ks*32 + (i<8 ? i : i+8) + (lane/16)*8
__global__ void pack_b_kernel(const float* __restrict__ B, _Float16* __restrict__ Bp,
                              int K, int Nc) {
  const int nTiles = (Nc + 15) >> 4;
  const int nSlab  = K >> 5;
  const long long total = (long long)nTiles * nSlab * 32 * 16;
  long long t = (long long)blockIdx.x * blockDim.x + threadIdx.x;
  if (t >= total) return;
  const int i    = (int)(t & 15);
  const int lane = (int)((t >> 4) & 31);
  const long long rest = t >> 9;
  const int ks = (int)(rest % nSlab);
  const int tn = (int)(rest / nSlab);
  const int lmod = lane & 15, half16 = lane >> 4;
  int col = tn * 16 + lmod;
  if (col >= Nc) col = Nc - 1;                    // clamped; masked at C store
  const int k = (ks << 5) + (i < 8 ? i : i + 8) + half16 * 8;
  Bp[t] = (_Float16)B[(size_t)k * Nc + col];
}

// ------------------- WMMA GEMM: C[M,Nc] = A[M,K] * Bp ---------------------
// A row-major f32 (converted to f16 on load, 4x b128 per fragment);
// B pre-packed f16 fragments (2x b128 contiguous per lane per fragment).
// One wave computes a 16 x (16*NT) strip; f32 accumulate.
template <int NT>
__global__ void wmma_gemm_kernel(const float* __restrict__ A,
                                 const _Float16* __restrict__ Bp,
                                 float* __restrict__ C,
                                 int M, int K, int Nc) {
  const int lane   = threadIdx.x & 31;
  const int wave   = threadIdx.x >> 5;
  const int lmod   = lane & 15;
  const int half16 = lane >> 4;

  const int nTiles  = (Nc + 15) >> 4;
  const int nGroups = (nTiles + NT - 1) / NT;
  const int mTiles  = (M + 15) >> 4;
  const long long wid = (long long)blockIdx.x * (blockDim.x >> 5) + wave;
  if (wid >= (long long)mTiles * nGroups) return;   // wave-uniform exit
  const int tm  = (int)(wid / nGroups);
  const int tn0 = (int)(wid % nGroups) * NT;

  int row = tm * 16 + lmod;
  if (row >= M) row = M - 1;               // clamp; padding rows masked at store
  const int kbase = half16 * 8;
  const int nSlab = K >> 5;

  v8f acc[NT];
#pragma unroll
  for (int t = 0; t < NT; ++t) acc[t] = (v8f){};

  const float* __restrict__ arow = A + (size_t)row * K;
#pragma unroll 2
  for (int ks = 0; ks < nSlab; ++ks) {
    // ---- B fragments first: NT x (2x b128) in one clause ----
    v16h bfrag[NT];
#pragma unroll
    for (int t = 0; t < NT; ++t)
      bfrag[t] = *(const v16h*)(Bp +
          ((((size_t)(tn0 + t)) * nSlab + ks) * 32 + lane) * 16);

    // ---- A fragment: two contiguous 8-float runs -> 4x float4 loads ----
    const float* ap = arow + (ks << 5) + kbase;
    const float4 a0 = *(const float4*)(ap + 0);
    const float4 a1 = *(const float4*)(ap + 4);
    const float4 a2 = *(const float4*)(ap + 16);
    const float4 a3 = *(const float4*)(ap + 20);
    v16h a;
    a[0]  = (_Float16)a0.x; a[1]  = (_Float16)a0.y;
    a[2]  = (_Float16)a0.z; a[3]  = (_Float16)a0.w;
    a[4]  = (_Float16)a1.x; a[5]  = (_Float16)a1.y;
    a[6]  = (_Float16)a1.z; a[7]  = (_Float16)a1.w;
    a[8]  = (_Float16)a2.x; a[9]  = (_Float16)a2.y;
    a[10] = (_Float16)a2.z; a[11] = (_Float16)a2.w;
    a[12] = (_Float16)a3.x; a[13] = (_Float16)a3.y;
    a[14] = (_Float16)a3.z; a[15] = (_Float16)a3.w;

#pragma unroll
    for (int t = 0; t < NT; ++t)
      acc[t] = __builtin_amdgcn_wmma_f32_16x16x32_f16(
          /*neg_a=*/false, a, /*neg_b=*/false, bfrag[t],
          /*c_mod=*/(short)0, acc[t], /*reuse_a=*/false, /*reuse_b=*/false);
  }

  // C/D layout: VGPR r holds M=r (lanes 0-15) / M=r+8 (lanes 16-31), N=lane%16
#pragma unroll
  for (int t = 0; t < NT; ++t) {
    const int col = (tn0 + t) * 16 + lmod;
    if (col < Nc) {
#pragma unroll
      for (int r = 0; r < 8; ++r) {
        const int orow = tm * 16 + r + half16 * 8;
        if (orow < M) C[(size_t)orow * Nc + col] = acc[t][r];
      }
    }
  }
}

// ---------- per-(node,head) attention logit halves: as, ad ----------------
__global__ void alpha_kernel(const float* __restrict__ h,
                             const float* __restrict__ att_s,
                             const float* __restrict__ att_d,
                             float* __restrict__ as_, float* __restrict__ ad_,
                             long long N, int H, int C) {
  long long t = (long long)blockIdx.x * blockDim.x + threadIdx.x;
  if (t >= N * H) return;
  const long long n = t / H;
  const int hh = (int)(t % H);
  const float* hp = h + (size_t)n * H * C + (size_t)hh * C;
  const float* sp = att_s + (size_t)hh * C;
  const float* dp = att_d + (size_t)hh * C;
  float s = 0.f, d = 0.f;
  for (int c = 0; c < C; ++c) {
    const float v = hp[c];
    s = fmaf(v, sp[c], s);
    d = fmaf(v, dp[c], d);
  }
  as_[t] = s;
  ad_[t] = d;
}

__device__ __forceinline__ void edge_sd(const long long* __restrict__ ei,
                                        long long E, long long e,
                                        long long& s, long long& d) {
  if (e < E) { s = ei[e]; d = ei[E + e]; }
  else       { s = d = e - E; }           // self-loops appended
}

// ------- segment-max over destination neighborhoods (1 thread/edge) -------
__global__ void edge_max_kernel(const long long* __restrict__ ei,
                                long long E, long long N,
                                const float* __restrict__ as_,
                                const float* __restrict__ ad_,
                                unsigned* __restrict__ m, int H) {
  const long long ET = E + N;
  long long e = (long long)blockIdx.x * blockDim.x + threadIdx.x;
  if (e >= ET) return;
  long long s, d;
  edge_sd(ei, E, e, s, d);
  for (int hh = 0; hh < H; ++hh) {
    float v = as_[s * H + hh] + ad_[d * H + hh];
    v = (v > 0.f) ? v : NEG_SLOPE * v;
    atomicMax(&m[d * H + hh], mapOrdered(v));
  }
}

// ------------ segment-sum of exp(e - max) (1 thread/edge) -----------------
__global__ void edge_sum_kernel(const long long* __restrict__ ei,
                                long long E, long long N,
                                const float* __restrict__ as_,
                                const float* __restrict__ ad_,
                                const unsigned* __restrict__ m,
                                float* __restrict__ den, int H) {
  const long long ET = E + N;
  long long e = (long long)blockIdx.x * blockDim.x + threadIdx.x;
  if (e >= ET) return;
  long long s, d;
  edge_sd(ei, E, e, s, d);
  for (int hh = 0; hh < H; ++hh) {
    float v = as_[s * H + hh] + ad_[d * H + hh];
    v = (v > 0.f) ? v : NEG_SLOPE * v;
    const float ex = __expf(v - unmapOrdered(m[d * H + hh]));
    atomicAdd(&den[d * H + hh], ex);
  }
}

// ---- weighted aggregation: out[dst] += h[src] * alpha   (1 wave / edge) --
__global__ void edge_agg_kernel(const long long* __restrict__ ei,
                                long long E, long long N,
                                const float* __restrict__ as_,
                                const float* __restrict__ ad_,
                                const unsigned* __restrict__ m,
                                const float* __restrict__ den,
                                const float* __restrict__ h,
                                float* __restrict__ out, int H, int C) {
  const int lane = threadIdx.x & 31;
  const int wave = threadIdx.x >> 5;
  const long long ET = E + N;
  const long long e = (long long)blockIdx.x * (blockDim.x >> 5) + wave;
  if (e >= ET) return;
  long long s, d;
  edge_sd(ei, E, e, s, d);
  // hoist per-head softmax weight (H <= 4)
  float alph[4];
#pragma unroll 4
  for (int hh = 0; hh < H; ++hh) {
    float v = as_[s * H + hh] + ad_[d * H + hh];
    v = (v > 0.f) ? v : NEG_SLOPE * v;
    const float ex = __expf(v - unmapOrdered(m[d * H + hh]));
    alph[hh] = ex / (den[d * H + hh] + 1e-16f);
  }
  const int HC = H * C;
  for (int c = lane; c < HC; c += 32)
    atomicAdd(&out[(size_t)d * HC + c], h[(size_t)s * HC + c] * alph[c / C]);
}

// ------------------ out = elu(agg + bias), in place -----------------------
__global__ void elu_bias_kernel(float* __restrict__ agg,
                                const float* __restrict__ b,
                                long long N, int HC) {
  long long i = (long long)blockIdx.x * blockDim.x + threadIdx.x;
  if (i >= N * HC) return;
  const float v = agg[i] + b[i % HC];
  agg[i] = (v > 0.f) ? v : expm1f(v);
}

// ---------------- log_softmax((agg2 + b2), axis=1); 1 warp / node ---------
__global__ void logsoftmax_kernel(const float* __restrict__ agg2,
                                  const float* __restrict__ b2,
                                  float* __restrict__ out,
                                  long long N, int C) {
  const int lane = threadIdx.x & 31;
  const int wave = threadIdx.x >> 5;
  const long long n = (long long)blockIdx.x * (blockDim.x >> 5) + wave;
  if (n >= N) return;
  const float NEG_INF = -__builtin_inff();
  float x0 = NEG_INF, x1 = NEG_INF;
  if (lane < C)      x0 = agg2[(size_t)n * C + lane] + b2[lane];
  if (lane + 32 < C) x1 = agg2[(size_t)n * C + lane + 32] + b2[lane + 32];
  float mx = fmaxf(x0, x1);
  for (int o = 16; o > 0; o >>= 1) mx = fmaxf(mx, __shfl_xor(mx, o, 32));
  float sm = 0.f;
  if (lane < C)      sm += __expf(x0 - mx);
  if (lane + 32 < C) sm += __expf(x1 - mx);
  for (int o = 16; o > 0; o >>= 1) sm += __shfl_xor(sm, o, 32);
  const float lse = __logf(sm) + mx;
  if (lane < C)      out[(size_t)n * C + lane] = x0 - lse;
  if (lane + 32 < C) out[(size_t)n * C + lane + 32] = x1 - lse;
}

// --------------------------------------------------------------------------
extern "C" void kernel_launch(void* const* d_in, const int* in_sizes, int n_in,
                              void* d_out, int out_size, void* d_ws, size_t ws_size,
                              hipStream_t stream) {
  const float*     x      = (const float*)d_in[0];
  const long long* ei     = (const long long*)d_in[1];   // int64 edge_index [2,E]
  const float*     W1     = (const float*)d_in[2];
  const float*     att_s1 = (const float*)d_in[3];
  const float*     att_d1 = (const float*)d_in[4];
  const float*     b1     = (const float*)d_in[5];
  const float*     W2     = (const float*)d_in[6];
  const float*     att_s2 = (const float*)d_in[7];
  const float*     att_d2 = (const float*)d_in[8];
  const float*     b2     = (const float*)d_in[9];

  const int  IN  = 256;
  const int  H1  = 4, C1 = 64, HC1 = 256;
  const int  C2  = 40;
  const long long N  = in_sizes[0] / IN;
  const long long E  = in_sizes[1] / 2;
  const long long ET = E + N;                    // with self-loops

  // ------- workspace layout (floats); layer-2 scratch aliases dead h1 -----
  float*    h1   = (float*)d_ws;                 // N*256
  float*    as1  = h1  + (size_t)N * HC1;        // N*4
  float*    ad1  = as1 + (size_t)N * H1;         // N*4
  unsigned* m1   = (unsigned*)(ad1 + (size_t)N * H1);   // N*4
  float*    den1 = (float*)(m1 + (size_t)N * H1);       // N*4
  float*    agg1 = den1 + (size_t)N * H1;        // N*256 (becomes elu(agg1+b1))
  // packed-f16 weight fragments (32B aligned: all offsets are mult. of 32B)
  _Float16* Bp1  = (_Float16*)(agg1 + (size_t)N * HC1); // 16*8*512 = 65536 halfs
  _Float16* Bp2  = Bp1 + 16 * 8 * 512;                  // 3*8*512  = 12288 halfs
  // layer 2 reuses h1's region: needs N*(2*40+4) = N*84 < N*256 floats
  float*    h2   = h1;                           // N*40
  float*    as2  = h2  + (size_t)N * C2;         // N
  float*    ad2  = as2 + N;                      // N
  unsigned* m2   = (unsigned*)(ad2 + N);         // N
  float*    den2 = (float*)(m2 + N);             // N
  float*    agg2 = den2 + N;                     // N*40

  const int TPB = 256;
  auto blks = [](long long work, int per) -> unsigned {
    return (unsigned)((work + per - 1) / per);
  };

  // ============================ layer 1 ====================================
  pack_b_kernel<<<blks((long long)16 * 8 * 512, TPB), TPB, 0, stream>>>(W1, Bp1, IN, HC1);
  {
    const long long waves = ((N + 15) / 16) * (((HC1 + 15) / 16) / 4); // NT=4
    wmma_gemm_kernel<4><<<blks(waves, 4), 128, 0, stream>>>(x, Bp1, h1, (int)N, IN, HC1);
  }
  alpha_kernel<<<blks(N * H1, TPB), TPB, 0, stream>>>(h1, att_s1, att_d1, as1, ad1, N, H1, C1);
  fill_u32_kernel<<<blks(N * H1, TPB), TPB, 0, stream>>>(m1, MAPPED_NEG_INF, N * H1);
  // den1 and agg1 are contiguous -> one zero fill
  fill_u32_kernel<<<blks(N * H1 + N * HC1, TPB), TPB, 0, stream>>>(
      (unsigned*)den1, 0u, N * H1 + N * HC1);
  edge_max_kernel<<<blks(ET, TPB), TPB, 0, stream>>>(ei, E, N, as1, ad1, m1, H1);
  edge_sum_kernel<<<blks(ET, TPB), TPB, 0, stream>>>(ei, E, N, as1, ad1, m1, den1, H1);
  edge_agg_kernel<<<blks(ET, 8), TPB, 0, stream>>>(ei, E, N, as1, ad1, m1, den1,
                                                   h1, agg1, H1, C1);
  elu_bias_kernel<<<blks(N * HC1, TPB), TPB, 0, stream>>>(agg1, b1, N, HC1);

  // ============================ layer 2 ====================================
  pack_b_kernel<<<blks((long long)3 * 8 * 512, TPB), TPB, 0, stream>>>(W2, Bp2, HC1, C2);
  {
    const long long waves = ((N + 15) / 16) * 1;  // NT=3 covers all 3 n-tiles
    wmma_gemm_kernel<3><<<blks(waves, 4), 128, 0, stream>>>(agg1, Bp2, h2, (int)N, HC1, C2);
  }
  alpha_kernel<<<blks(N, TPB), TPB, 0, stream>>>(h2, att_s2, att_d2, as2, ad2, N, 1, C2);
  fill_u32_kernel<<<blks(N, TPB), TPB, 0, stream>>>(m2, MAPPED_NEG_INF, N);
  // den2 and agg2 contiguous -> one zero fill
  fill_u32_kernel<<<blks(N + N * C2, TPB), TPB, 0, stream>>>((unsigned*)den2, 0u, N + N * C2);
  edge_max_kernel<<<blks(ET, TPB), TPB, 0, stream>>>(ei, E, N, as2, ad2, m2, 1);
  edge_sum_kernel<<<blks(ET, TPB), TPB, 0, stream>>>(ei, E, N, as2, ad2, m2, den2, 1);
  edge_agg_kernel<<<blks(ET, 8), TPB, 0, stream>>>(ei, E, N, as2, ad2, m2, den2,
                                                   h2, agg2, 1, C2);
  logsoftmax_kernel<<<blks(N, 8), TPB, 0, stream>>>(agg2, b2, (float*)d_out, N, C2);
}
